// Network_68968584839826
// MI455X (gfx1250) — compile-verified, matrix operand order checked by reading, not run on hardware
//
#include <hip/hip_runtime.h>

// ---------------------------------------------------------------------------
// Fused 2-layer RBF network for MI455X (gfx1250, wave32, WMMA).
//   dist^2 = ||h||^2 + ||c||^2 - 2 h.c   -> GEMM + elementwise exp
// All four GEMM stages use v_wmma_f32_16x16x32_bf16 (K=32 matches widths).
// ---------------------------------------------------------------------------

typedef __attribute__((ext_vector_type(16))) __bf16 v16bf;
typedef __attribute__((ext_vector_type(8)))  float  v8f;

#define WAVES          8
#define ROWS_PER_BLOCK (WAVES * 16)

#define D0 32    // input width
#define C0 512   // layer-0 centers
#define H1 64    // hidden width
#define C1 512   // layer-1 centers
#define D2 32    // output width
#define NROWS 8192

__device__ __forceinline__ float sumsq4(float4 v) {
    return v.x * v.x + v.y * v.y + v.z * v.z + v.w * v.w;
}

// Build a v16bf A/B fragment from two contiguous 8-float chunks (a,b | c,d).
__device__ __forceinline__ v16bf pack16(float4 a, float4 b, float4 c, float4 d) {
    v16bf v;
    v[0]  = (__bf16)a.x; v[1]  = (__bf16)a.y; v[2]  = (__bf16)a.z; v[3]  = (__bf16)a.w;
    v[4]  = (__bf16)b.x; v[5]  = (__bf16)b.y; v[6]  = (__bf16)b.z; v[7]  = (__bf16)b.w;
    v[8]  = (__bf16)c.x; v[9]  = (__bf16)c.y; v[10] = (__bf16)c.z; v[11] = (__bf16)c.w;
    v[12] = (__bf16)d.x; v[13] = (__bf16)d.y; v[14] = (__bf16)d.z; v[15] = (__bf16)d.w;
    return v;
}

__device__ __forceinline__ v8f wmma_bf16(v16bf a, v16bf b, v8f c) {
    return __builtin_amdgcn_wmma_f32_16x16x32_bf16(
        /*neg_a=*/false, a, /*neg_b=*/false, b,
        /*c_mod=*/(short)0, c, /*reuse_a=*/false, /*reuse_b=*/false);
}

__global__ __launch_bounds__(WAVES * 32, 1)
void rbfnet2_fused_kernel(const float* __restrict__ x,
                          const float* __restrict__ c0, const float* __restrict__ ls0,
                          const float* __restrict__ W0, const float* __restrict__ b0,
                          const float* __restrict__ c1, const float* __restrict__ ls1,
                          const float* __restrict__ W1, const float* __restrict__ b1,
                          float* __restrict__ out)
{
    // Wave-private LDS scratch (no cross-wave sharing -> no block barriers).
    __shared__ __align__(16) float sPhi[WAVES][16 * 32];  // phi bounce D->A (2 KB/wave)
    __shared__ __align__(16) float sH[WAVES][16 * H1];    // hidden row block (4 KB/wave)
    __shared__ float sHN[WAVES][16];                      // row norms

    const int tid  = threadIdx.x;
    const int wave = tid >> 5;
    const int lane = tid & 31;
    const int n    = lane & 15;  // column within 16-wide tile (B/C/D layouts)
    const int mrow = lane & 15;  // row within 16-row tile   (A layout)
    const int hi   = lane >> 4;  // half selector: K chunks {hi*8.., hi*8+16..}
    const int row0 = blockIdx.x * ROWS_PER_BLOCK + wave * 16;

    float* __restrict__ phiW = sPhi[wave];
    float* __restrict__ hW   = sH[wave];
    float* __restrict__ hnW  = sHN[wave];

    // ---- stage A0: load x tile (16x32) into A-layout fragment + row norms ----
    const float* xr = x + (size_t)(row0 + mrow) * D0 + hi * 8;
    float4 xa = *(const float4*)(xr);
    float4 xb = *(const float4*)(xr + 4);
    float4 xc = *(const float4*)(xr + 16);
    float4 xd = *(const float4*)(xr + 20);
    v16bf a0 = pack16(xa, xb, xc, xd);
    float xn = sumsq4(xa) + sumsq4(xb) + sumsq4(xc) + sumsq4(xd);
    xn += __shfl_xor(xn, 16, 32);             // lanes l, l^16 hold row halves
    if (lane < 16) hnW[mrow] = xn;
    __builtin_amdgcn_wave_barrier();

    // Per-lane D-layout row norms: VGPR r of D holds row (r + 8*hi).
    float hnv[8];
#pragma unroll
    for (int r = 0; r < 8; ++r) hnv[r] = hnW[r + 8 * hi];

    // Bias preloads (per-lane column n of each 16-wide output tile).
    float b0v[4];
#pragma unroll
    for (int nt = 0; nt < 4; ++nt) b0v[nt] = b0[nt * 16 + n];
    float b1v[2];
#pragma unroll
    for (int nt = 0; nt < 2; ++nt) b1v[nt] = b1[nt * 16 + n];

    // =========================== Layer 0 ===================================
    v8f hacc[4];
#pragma unroll
    for (int nt = 0; nt < 4; ++nt) hacc[nt] = (v8f){0, 0, 0, 0, 0, 0, 0, 0};

    for (int ct = 0; ct < C0 / 16; ++ct) {
        __builtin_prefetch(c0 + (size_t)((ct + 1) & 31) * 16 * D0, 0, 3);

        // B fragment = centers0 tile (K=32, N=16); norm for free via shfl.
        const float* cr = c0 + (size_t)(ct * 16 + n) * D0 + hi * 8;
        float4 ca = *(const float4*)(cr);
        float4 cb = *(const float4*)(cr + 4);
        float4 cc = *(const float4*)(cr + 16);
        float4 cd = *(const float4*)(cr + 20);
        v16bf bfrag = pack16(ca, cb, cc, cd);
        float cn = sumsq4(ca) + sumsq4(cb) + sumsq4(cc) + sumsq4(cd);
        cn += __shfl_xor(cn, 16, 32);
        float is = __expf(-2.f * ls0[ct * 16 + n]);   // 1/sigma^2

        v8f z = (v8f){0, 0, 0, 0, 0, 0, 0, 0};
        v8f d = wmma_bf16(a0, bfrag, z);              // d = x . c  (16x16)

        // phi = exp(-(||x||^2 + ||c||^2 - 2 d) / sigma^2), bounce to LDS.
#pragma unroll
        for (int r = 0; r < 8; ++r) {
            float dist2 = hnv[r] + cn - 2.f * d[r];
            phiW[(r + 8 * hi) * 32 + (ct & 1) * 16 + n] = __expf(-dist2 * is);
        }
        __builtin_amdgcn_wave_barrier();

        if (ct & 1) {  // have a full K=32 slab of phi -> GEMM2 with W0
            const float* pr = phiW + mrow * 32 + hi * 8;
            float4 p0 = *(const float4*)(pr);
            float4 p1 = *(const float4*)(pr + 4);
            float4 p2 = *(const float4*)(pr + 16);
            float4 p3 = *(const float4*)(pr + 20);
            v16bf a2 = pack16(p0, p1, p2, p3);
            const int kbase = (ct - 1) * 16;
#pragma unroll
            for (int nt = 0; nt < 4; ++nt) {
                const float* wr = W0 + (size_t)(nt * 16 + n) * C0 + kbase + hi * 8;
                float4 w0 = *(const float4*)(wr);
                float4 w1v = *(const float4*)(wr + 4);
                float4 w2 = *(const float4*)(wr + 16);
                float4 w3 = *(const float4*)(wr + 20);
                v16bf bw = pack16(w0, w1v, w2, w3);
                hacc[nt] = wmma_bf16(a2, bw, hacc[nt]);
            }
            __builtin_amdgcn_wave_barrier();
        }
    }

    // ---- h = phi0 @ W0^T + b0 -> LDS; recompute row norms; build A1 frags ----
#pragma unroll
    for (int nt = 0; nt < 4; ++nt)
#pragma unroll
        for (int r = 0; r < 8; ++r)
            hW[(r + 8 * hi) * H1 + nt * 16 + n] = hacc[nt][r] + b0v[nt];
    __builtin_amdgcn_wave_barrier();

    if (lane < 16) {
        float s = 0.f;
        const float4* hr = (const float4*)(hW + mrow * H1);
#pragma unroll
        for (int q = 0; q < H1 / 4; ++q) s += sumsq4(hr[q]);
        hnW[mrow] = s;
    }
    __builtin_amdgcn_wave_barrier();
#pragma unroll
    for (int r = 0; r < 8; ++r) hnv[r] = hnW[r + 8 * hi];

    v16bf a1[2];
#pragma unroll
    for (int kk = 0; kk < 2; ++kk) {
        const float* hr = hW + mrow * H1 + kk * 32 + hi * 8;
        float4 h0 = *(const float4*)(hr);
        float4 h1 = *(const float4*)(hr + 4);
        float4 h2 = *(const float4*)(hr + 16);
        float4 h3 = *(const float4*)(hr + 20);
        a1[kk] = pack16(h0, h1, h2, h3);
    }

    // =========================== Layer 1 ===================================
    v8f oacc[2];
#pragma unroll
    for (int nt = 0; nt < 2; ++nt) oacc[nt] = (v8f){0, 0, 0, 0, 0, 0, 0, 0};

    for (int ct = 0; ct < C1 / 16; ++ct) {
        __builtin_prefetch(c1 + (size_t)((ct + 1) & 31) * 16 * H1, 0, 3);

        float cn = 0.f;
        v16bf bf[2];
#pragma unroll
        for (int kk = 0; kk < 2; ++kk) {
            const float* cr = c1 + (size_t)(ct * 16 + n) * H1 + kk * 32 + hi * 8;
            float4 ca = *(const float4*)(cr);
            float4 cb = *(const float4*)(cr + 4);
            float4 cc = *(const float4*)(cr + 16);
            float4 cd = *(const float4*)(cr + 20);
            bf[kk] = pack16(ca, cb, cc, cd);
            cn += sumsq4(ca) + sumsq4(cb) + sumsq4(cc) + sumsq4(cd);
        }
        cn += __shfl_xor(cn, 16, 32);
        float is = __expf(-2.f * ls1[ct * 16 + n]);

        v8f z = (v8f){0, 0, 0, 0, 0, 0, 0, 0};
        v8f d = wmma_bf16(a1[0], bf[0], z);           // K = 0..31
        d = wmma_bf16(a1[1], bf[1], d);               // K = 32..63

#pragma unroll
        for (int r = 0; r < 8; ++r) {
            float dist2 = hnv[r] + cn - 2.f * d[r];
            phiW[(r + 8 * hi) * 32 + (ct & 1) * 16 + n] = __expf(-dist2 * is);
        }
        __builtin_amdgcn_wave_barrier();

        if (ct & 1) {  // GEMM4 with W1
            const float* pr = phiW + mrow * 32 + hi * 8;
            float4 p0 = *(const float4*)(pr);
            float4 p1 = *(const float4*)(pr + 4);
            float4 p2 = *(const float4*)(pr + 16);
            float4 p3 = *(const float4*)(pr + 20);
            v16bf a2 = pack16(p0, p1, p2, p3);
            const int kbase = (ct - 1) * 16;
#pragma unroll
            for (int nt = 0; nt < 2; ++nt) {
                const float* wr = W1 + (size_t)(nt * 16 + n) * C1 + kbase + hi * 8;
                float4 w0 = *(const float4*)(wr);
                float4 w1v = *(const float4*)(wr + 4);
                float4 w2 = *(const float4*)(wr + 16);
                float4 w3 = *(const float4*)(wr + 20);
                v16bf bw = pack16(w0, w1v, w2, w3);
                oacc[nt] = wmma_bf16(a2, bw, oacc[nt]);
            }
            __builtin_amdgcn_wave_barrier();
        }
    }

    // ---- out = phi1 @ W1^T + b1 (D layout -> row-major global) ----
#pragma unroll
    for (int nt = 0; nt < 2; ++nt)
#pragma unroll
        for (int r = 0; r < 8; ++r)
            out[(size_t)(row0 + r + 8 * hi) * D2 + nt * 16 + n] = oacc[nt][r] + b1v[nt];
}

extern "C" void kernel_launch(void* const* d_in, const int* in_sizes, int n_in,
                              void* d_out, int out_size, void* d_ws, size_t ws_size,
                              hipStream_t stream) {
    (void)in_sizes; (void)n_in; (void)out_size; (void)d_ws; (void)ws_size;
    const float* x   = (const float*)d_in[0];
    const float* c0  = (const float*)d_in[1];
    const float* ls0 = (const float*)d_in[2];
    const float* W0  = (const float*)d_in[3];
    const float* b0  = (const float*)d_in[4];
    const float* c1  = (const float*)d_in[5];
    const float* ls1 = (const float*)d_in[6];
    const float* W1  = (const float*)d_in[7];
    const float* b1  = (const float*)d_in[8];
    float* out = (float*)d_out;

    dim3 grid(NROWS / ROWS_PER_BLOCK);   // 64 blocks
    dim3 block(WAVES * 32);              // 256 threads = 8 waves
    rbfnet2_fused_kernel<<<grid, block, 0, stream>>>(
        x, c0, ls0, W0, b0, c1, ls1, W1, b1, out);
}
